// SecondOrderDeformableAlignment_61014305407378
// MI455X (gfx1250) — matrix operand order repaired and checked by reading, not compiled, not run on hardware
//
#include <hip/hip_runtime.h>

// ---------------------------------------------------------------------------
// MI455X (gfx1250, wave32) implementation of SecondOrderDeformableAlignment.
// All convs + the deformable-conv GEMM run on v_wmma_f32_16x16x32_f16
// (f16 inputs, f32 accumulation). Bilinear gather / tanh / sigmoid in f32.
// Each wave computes a 16-och x 32-px tile (2 accumulators) so the B (weight)
// fragment is reused across two WMMAs -> 3 b128 loads per WMMA instead of 4.
// ---------------------------------------------------------------------------

typedef _Float16 v16h __attribute__((ext_vector_type(16)));
typedef float    v8f  __attribute__((ext_vector_type(8)));

union Frag16 { v16h v; int4 q[2]; };   // 32 bytes: 16 halves
union St8h   { int4 q; _Float16 h[8]; };

#define HW_H 128
#define HW_W 128
#define HWPX (HW_H * HW_W)

// ---------------------------------------------------------------------------
// Weight repack: w[Cout][Cin][3][3] f32  ->  wt[t=9][Cout][Cin_pad] f16
// ---------------------------------------------------------------------------
__global__ void repack_w_kernel(const float* __restrict__ w, _Float16* __restrict__ wt,
                                int Cout, int Cin, int CP) {
    int i = blockIdx.x * 256 + threadIdx.x;
    int total = 9 * Cout * CP;
    if (i >= total) return;
    int ci = i % CP;
    int oc = (i / CP) % Cout;
    int t  = i / (CP * Cout);
    float v = (ci < Cin) ? w[(oc * Cin + ci) * 9 + t] : 0.0f;
    wt[((size_t)t * Cout + oc) * CP + ci] = (_Float16)v;
}

// ---------------------------------------------------------------------------
// Generic 3x3 conv (stride 1, pad 1) as implicit GEMM.
// Block: 128 threads = 4 waves. Block tile: 32-pixel strip x 64 och.
// Each wave: 16 och x 32 px via two accumulators sharing the B fragment.
// A: channel-interleaved LDS halo patch [3][34][Cin_pad(+8)].
// B: repacked global f16 weights.
// ---------------------------------------------------------------------------
template <typename TIn, typename TOut, bool LRELU>
__global__ void conv3x3_wmma_kernel(const TIn* __restrict__ in,
                                    const _Float16* __restrict__ wt,
                                    const float* __restrict__ bias,
                                    TOut* __restrict__ out,
                                    int Cin, int CP, int Cout) {
    extern __shared__ __align__(16) char smem_raw[];
    _Float16* lds = (_Float16*)smem_raw;

    const int tid  = threadIdx.x;
    const int lane = tid & 31;
    const int wave = tid >> 5;
    const int b    = blockIdx.z;
    const int strip = blockIdx.x;            // 512 strips of 32 px per image
    const int y  = strip >> 2;               // W/32 == 4
    const int x0 = (strip & 3) << 5;
    const int rowstride = CP + 8;            // pad to spread LDS banks

    // ---- stage halo patch [3 rows][34 cols][Cin_pad] as f16 ----
    const int total = 3 * 34 * CP;
    for (int i = tid; i < total; i += 128) {
        int ci = i % CP;
        int rc = i / CP;
        int c  = rc % 34;
        int r  = rc / 34;
        int gy = y + r - 1;
        int gx = x0 + c - 1;
        float v = 0.0f;
        if (ci < Cin && gy >= 0 && gy < HW_H && gx >= 0 && gx < HW_W)
            v = (float)in[((size_t)(b * Cin + ci)) * HWPX + gy * HW_W + gx];
        lds[(r * 34 + c) * rowstride + ci] = (_Float16)v;
    }
    __syncthreads();

    const int ochb = blockIdx.y * 64 + wave * 16;
    if (ochb >= Cout) return;                // tail tile guard (Cout=432)

    const int m    = lane & 15;              // A: M = pixel-in-strip (low half)
    const int och  = ochb + m;               // B: N = output channel
    const int kbA  = (lane < 16) ? 0 : 8;    // ISA A layout: K 0-7/16-23 vs 8-15/24-31
    const int kbB  = (lane < 16) ? 0 : 8;    // B layout: K 0-7/16-23 vs 8-15/24-31
    const int nch  = CP >> 5;

    v8f acc0 = {};
    v8f acc1 = {};
    for (int t = 0; t < 9; ++t) {
        const int ky = t / 3, kx = t % 3;
        const _Float16* arow0 = &lds[(ky * 34 + m + kx) * rowstride];
        const _Float16* arow1 = arow0 + 16 * rowstride;   // pixels 16..31
        const _Float16* brow  = &wt[((size_t)t * Cout + och) * CP];
        for (int c32 = 0; c32 < nch; ++c32) {
            Frag16 fb;
            const int4* pb = (const int4*)(brow + c32 * 32 + kbB);
            fb.q[0] = pb[0];
            fb.q[1] = pb[2];
            Frag16 fa0, fa1;
            const int4* pa0 = (const int4*)(arow0 + c32 * 32 + kbA);
            fa0.q[0] = pa0[0]; fa0.q[1] = pa0[2];
            const int4* pa1 = (const int4*)(arow1 + c32 * 32 + kbA);
            fa1.q[0] = pa1[0]; fa1.q[1] = pa1[2];
            acc0 = __builtin_amdgcn_wmma_f32_16x16x32_f16(
                false, fa0.v, false, fb.v, (short)0, acc0, false, false);
            acc1 = __builtin_amdgcn_wmma_f32_16x16x32_f16(
                false, fa1.v, false, fb.v, (short)0, acc1, false, false);
        }
    }

    // ---- epilogue: bias (+ leaky ReLU), 8 consecutive-x values per store ----
    const float bv  = bias[och];
    const int   px0 = (lane < 16) ? 0 : 8;   // D layout: lanes 0-15 M=r, 16-31 M=8+r
    const size_t ob0 = ((size_t)(b * Cout + och)) * HWPX + (size_t)y * HW_W + x0 + px0;
    const size_t ob1 = ob0 + 16;

    if constexpr (sizeof(TOut) == 2) {
        St8h s0, s1;
#pragma unroll
        for (int r = 0; r < 8; ++r) {
            float v0 = acc0[r] + bv;
            float v1 = acc1[r] + bv;
            if (LRELU) {
                v0 = (v0 >= 0.0f) ? v0 : 0.1f * v0;
                v1 = (v1 >= 0.0f) ? v1 : 0.1f * v1;
            }
            s0.h[r] = (_Float16)v0;
            s1.h[r] = (_Float16)v1;
        }
        *(int4*)((_Float16*)out + ob0) = s0.q;
        *(int4*)((_Float16*)out + ob1) = s1.q;
    } else {
        float v0[8], v1[8];
#pragma unroll
        for (int r = 0; r < 8; ++r) {
            float a = acc0[r] + bv;
            float c = acc1[r] + bv;
            if (LRELU) {
                a = (a >= 0.0f) ? a : 0.1f * a;
                c = (c >= 0.0f) ? c : 0.1f * c;
            }
            v0[r] = a; v1[r] = c;
        }
        *(float4*)((float*)out + ob0)     = make_float4(v0[0], v0[1], v0[2], v0[3]);
        *(float4*)((float*)out + ob0 + 4) = make_float4(v0[4], v0[5], v0[6], v0[7]);
        *(float4*)((float*)out + ob1)     = make_float4(v1[0], v1[1], v1[2], v1[3]);
        *(float4*)((float*)out + ob1 + 4) = make_float4(v1[4], v1[5], v1[6], v1[7]);
    }
}

// ---------------------------------------------------------------------------
// Modulated deformable conv. Block: 128 threads, one 32-pixel strip, 64 och.
// Phase 1: build cols[32 px][9 taps][64 cin] f16 in LDS — fuses
//   offset = 10*tanh(raw) + flow[::-1],  mask = sigmoid(raw),  bilinear gather.
// Phase 2: WMMA GEMM (2 px-tiles per wave, B reused), + bias, f32 store.
// ---------------------------------------------------------------------------
__global__ void deform_wmma_kernel(const float* __restrict__ x,
                                   const float* __restrict__ c4,     // raw conv4 out [432][H][W]
                                   const float* __restrict__ flow,
                                   const _Float16* __restrict__ wt,  // [9][64][64] f16
                                   const float* __restrict__ bias,
                                   float* __restrict__ out) {
    const int ROWS = 584;                    // 9*64 + 8 pad halves
    __shared__ __align__(16) _Float16 cols[32 * ROWS];

    const int tid  = threadIdx.x;
    const int lane = tid & 31;
    const int wave = tid >> 5;
    const int b    = blockIdx.z;
    const int strip = blockIdx.x;            // 512 strips of 32 px
    const int y  = strip >> 2;
    const int x0 = (strip & 3) << 5;

    const float* xb  = x    + (size_t)b * 64  * HWPX;
    const float* c4b = c4   + (size_t)b * 432 * HWPX;
    const float* flb = flow + (size_t)b * 2   * HWPX;

    // ---- phase 1: 32 px * 16 groups * 9 taps tasks, 4 channels each ----
    for (int it = tid; it < 32 * 16 * 9; it += 128) {
        const int px  = it / 144;
        const int rem = it % 144;
        const int dg  = rem / 9;
        const int t   = rem % 9;
        const int gx  = x0 + px;
        const int pix = y * HW_W + gx;
        const int ky = t / 3, kx = t % 3;

        const float dyr = c4b[(size_t)(((dg * 9 + t) * 2 + 0)) * HWPX + pix];
        const float dxr = c4b[(size_t)(((dg * 9 + t) * 2 + 1)) * HWPX + pix];
        const float mr  = c4b[(size_t)(288 + dg * 9 + t) * HWPX + pix];
        // flow[:, ::-1] tiled: even offset chan (dy) gets flow ch1, odd (dx) gets ch0
        const float dy = 10.0f * tanhf(dyr) + flb[1 * HWPX + pix];
        const float dx = 10.0f * tanhf(dxr) + flb[0 * HWPX + pix];
        const float msk = 1.0f / (1.0f + expf(-mr));

        const float py  = (float)(y + ky - 1) + dy;
        const float pxx = (float)(gx + kx - 1) + dx;
        const float fy = floorf(py), fx = floorf(pxx);
        const int y0i = (int)fy, x0i = (int)fx;
        const float wy = py - fy, wx = pxx - fx;

        const bool v0y = (y0i >= 0) && (y0i < HW_H);
        const bool v1y = (y0i + 1 >= 0) && (y0i + 1 < HW_H);
        const bool v0x = (x0i >= 0) && (x0i < HW_W);
        const bool v1x = (x0i + 1 >= 0) && (x0i + 1 < HW_W);
        const int yc0 = y0i < 0 ? 0 : (y0i > HW_H - 1 ? HW_H - 1 : y0i);
        const int yc1 = y0i + 1 < 0 ? 0 : (y0i + 1 > HW_H - 1 ? HW_H - 1 : y0i + 1);
        const int xc0 = x0i < 0 ? 0 : (x0i > HW_W - 1 ? HW_W - 1 : x0i);
        const int xc1 = x0i + 1 < 0 ? 0 : (x0i + 1 > HW_W - 1 ? HW_W - 1 : x0i + 1);
        const float f00 = (v0y && v0x) ? (1.0f - wy) * (1.0f - wx) : 0.0f;
        const float f01 = (v0y && v1x) ? (1.0f - wy) * wx : 0.0f;
        const float f10 = (v1y && v0x) ? wy * (1.0f - wx) : 0.0f;
        const float f11 = (v1y && v1x) ? wy * wx : 0.0f;

#pragma unroll
        for (int j = 0; j < 4; ++j) {
            const int cin = dg * 4 + j;
            const float* xc = xb + (size_t)cin * HWPX;
            float a = f00 * xc[yc0 * HW_W + xc0] + f01 * xc[yc0 * HW_W + xc1]
                    + f10 * xc[yc1 * HW_W + xc0] + f11 * xc[yc1 * HW_W + xc1];
            cols[px * ROWS + t * 64 + cin] = (_Float16)(a * msk);
        }
    }
    __syncthreads();

    // ---- phase 2: 64 och x 32 px GEMM over K = 9*64 ----
    const int m   = lane & 15;
    const int och = wave * 16 + m;
    const int kbA = (lane < 16) ? 0 : 8;
    const int kbB = (lane < 16) ? 0 : 8;

    v8f acc0 = {};
    v8f acc1 = {};
    for (int t = 0; t < 9; ++t) {
        const _Float16* arow0 = &cols[m * ROWS + t * 64];
        const _Float16* arow1 = arow0 + 16 * ROWS;
        const _Float16* brow  = &wt[((size_t)t * 64 + och) * 64];
#pragma unroll
        for (int c32 = 0; c32 < 2; ++c32) {
            Frag16 fb;
            const int4* pb = (const int4*)(brow + c32 * 32 + kbB);
            fb.q[0] = pb[0]; fb.q[1] = pb[2];
            Frag16 fa0, fa1;
            const int4* pa0 = (const int4*)(arow0 + c32 * 32 + kbA);
            fa0.q[0] = pa0[0]; fa0.q[1] = pa0[2];
            const int4* pa1 = (const int4*)(arow1 + c32 * 32 + kbA);
            fa1.q[0] = pa1[0]; fa1.q[1] = pa1[2];
            acc0 = __builtin_amdgcn_wmma_f32_16x16x32_f16(
                false, fa0.v, false, fb.v, (short)0, acc0, false, false);
            acc1 = __builtin_amdgcn_wmma_f32_16x16x32_f16(
                false, fa1.v, false, fb.v, (short)0, acc1, false, false);
        }
    }

    const float bv  = bias[och];
    const int   px0 = (lane < 16) ? 0 : 8;
    const size_t ob0 = ((size_t)(b * 64 + och)) * HWPX + (size_t)y * HW_W + x0 + px0;
    const size_t ob1 = ob0 + 16;
    *(float4*)(out + ob0)     = make_float4(acc0[0] + bv, acc0[1] + bv, acc0[2] + bv, acc0[3] + bv);
    *(float4*)(out + ob0 + 4) = make_float4(acc0[4] + bv, acc0[5] + bv, acc0[6] + bv, acc0[7] + bv);
    *(float4*)(out + ob1)     = make_float4(acc1[0] + bv, acc1[1] + bv, acc1[2] + bv, acc1[3] + bv);
    *(float4*)(out + ob1 + 4) = make_float4(acc1[4] + bv, acc1[5] + bv, acc1[6] + bv, acc1[7] + bv);
}

// ---------------------------------------------------------------------------
// Host orchestration
// ---------------------------------------------------------------------------
extern "C" void kernel_launch(void* const* d_in, const int* in_sizes, int n_in,
                              void* d_out, int out_size, void* d_ws, size_t ws_size,
                              hipStream_t stream) {
    (void)in_sizes; (void)n_in; (void)out_size; (void)ws_size;

    const float* x      = (const float*)d_in[0];
    const float* cond   = (const float*)d_in[1];
    const float* flow   = (const float*)d_in[2];
    const float* w1     = (const float*)d_in[3];
    const float* b1     = (const float*)d_in[4];
    const float* w2     = (const float*)d_in[5];
    const float* b2     = (const float*)d_in[6];
    const float* w3     = (const float*)d_in[7];
    const float* b3     = (const float*)d_in[8];
    const float* w4     = (const float*)d_in[9];
    const float* b4     = (const float*)d_in[10];
    const float* weight = (const float*)d_in[11];
    const float* bias   = (const float*)d_in[12];

    char* ws = (char*)d_ws;
    size_t off = 0;
    auto take = [&](size_t bytes) {
        size_t cur = off;
        off += (bytes + 255) & ~(size_t)255;
        return cur;
    };
    const size_t HW = (size_t)HWPX;
    _Float16* wt1 = (_Float16*)(ws + take((size_t)9 * 64  * 160 * 2));
    _Float16* wt2 = (_Float16*)(ws + take((size_t)9 * 64  * 64  * 2));
    _Float16* wt3 = (_Float16*)(ws + take((size_t)9 * 64  * 64  * 2));
    _Float16* wt4 = (_Float16*)(ws + take((size_t)9 * 432 * 64  * 2));
    _Float16* wt5 = (_Float16*)(ws + take((size_t)9 * 64  * 64  * 2));
    _Float16* h1  = (_Float16*)(ws + take((size_t)2 * 64  * HW * 2));
    _Float16* h2  = (_Float16*)(ws + take((size_t)2 * 64  * HW * 2));
    _Float16* h3  = (_Float16*)(ws + take((size_t)2 * 64  * HW * 2));
    float*    c4  = (float*)   (ws + take((size_t)2 * 432 * HW * 4));

    auto cdiv = [](int a, int b) { return (a + b - 1) / b; };

    // weight repacks (f32 OIHW -> f16 [tap][Cout][Cin_pad])
    repack_w_kernel<<<cdiv(9 * 64  * 160, 256), 256, 0, stream>>>(w1,     wt1, 64,  133, 160);
    repack_w_kernel<<<cdiv(9 * 64  * 64,  256), 256, 0, stream>>>(w2,     wt2, 64,  64,  64);
    repack_w_kernel<<<cdiv(9 * 64  * 64,  256), 256, 0, stream>>>(w3,     wt3, 64,  64,  64);
    repack_w_kernel<<<cdiv(9 * 432 * 64,  256), 256, 0, stream>>>(w4,     wt4, 432, 64,  64);
    repack_w_kernel<<<cdiv(9 * 64  * 64,  256), 256, 0, stream>>>(weight, wt5, 64,  64,  64);

    const size_t lds133 = (size_t)3 * 34 * (160 + 8) * 2;   // 34272 B
    const size_t lds64  = (size_t)3 * 34 * (64 + 8)  * 2;   // 14688 B
    const dim3 grid1(512, 1, 2);                            // 32-px strips
    const dim3 grid4(512, 7, 2);                            // ceil(432/64)=7, tail guarded

    conv3x3_wmma_kernel<float,    _Float16, true ><<<grid1, 128, lds133, stream>>>(cond, wt1, b1, h1, 133, 160, 64);
    conv3x3_wmma_kernel<_Float16, _Float16, true ><<<grid1, 128, lds64,  stream>>>(h1,   wt2, b2, h2, 64,  64,  64);
    conv3x3_wmma_kernel<_Float16, _Float16, true ><<<grid1, 128, lds64,  stream>>>(h2,   wt3, b3, h3, 64,  64,  64);
    conv3x3_wmma_kernel<_Float16, float,    false><<<grid4, 128, lds64,  stream>>>(h3,   wt4, b4, c4, 64,  64,  432);

    deform_wmma_kernel<<<grid1, 128, 0, stream>>>(x, c4, flow, wt5, bias, (float*)d_out);
}
